// LFModel_23476291240387
// MI455X (gfx1250) — matrix operand order repaired and checked by reading, not compiled, not run on hardware
//
#include <hip/hip_runtime.h>
#include <hip/hip_bf16.h>
#include <math.h>

// ---------------------------------------------------------------------------
// Problem constants (match reference)
// ---------------------------------------------------------------------------
#define NTOK   4096      // B*S
#define D_     256
#define DLF_   768
#define ADAPT_ 128
#define NEXP_  4
#define HID_   64
#define DSUM_  1536      // 3*D + D_LF

typedef __attribute__((ext_vector_type(16))) __bf16 v16bf;
typedef __attribute__((ext_vector_type(8)))  __bf16 v8bf;
typedef __attribute__((ext_vector_type(4)))  __bf16 v4bf;
typedef __attribute__((ext_vector_type(8)))  float  v8f;

__device__ __forceinline__ float gelu_exact(float v) {
    return 0.5f * v * (1.0f + erff(v * 0.70710678118654752f));
}

// 16B global -> LDS async copy (CDNA5 TDM-lite path, ASYNCcnt-tracked)
__device__ __forceinline__ void async_copy_b128(unsigned ldsOff, const void* gaddr) {
    asm volatile("global_load_async_to_lds_b128 %0, %1, off"
                 :: "v"(ldsOff), "v"((unsigned long long)(uintptr_t)gaddr)
                 : "memory");
}
__device__ __forceinline__ void wait_async0() {
    asm volatile("s_wait_asynccnt 0x0" ::: "memory");
}

// ---------------------------------------------------------------------------
// Tiled WMMA GEMM:  C[M,N] = act(A[M,K] @ B[K,N] + bias[N])
//   Requirements (guaranteed by all launches): M%128==0, N%BN==0, K%32==0.
//   * A: bf16 row-major [M,K]; Bt: bf16 PRE-TRANSPOSED [N,K]
//   * tiles staged to LDS with global_load_async_to_lds_b128, double-buffered:
//     copies for tile i+1 overlap WMMAs on tile i; s_wait_asynccnt + barrier
//   * 256 threads = 8 wave32 (4 along M x 2 along N); block tile 128 x BN
//   * each wave: 32 x (BN/2) macro-tile = 2 x (BN/32) WMMA 16x16x32 bf16
//   * FUSED_PHI: A(m,k) = gelu(x_bf16[m, k>>6] * Wphi[k] + bphi[k]) (KA expert)
// ---------------------------------------------------------------------------
template<int ACT, bool FUSED_PHI, int BN, bool OUT_BF16>
__global__ __launch_bounds__(256)
void gemm_wmma_bf16(const __bf16* __restrict__ A, const __bf16* __restrict__ Bt,
                    const float* __restrict__ bias, void* __restrict__ Cv,
                    int M, int N, int K,
                    const float* __restrict__ Wphi, const float* __restrict__ bphi)
{
    constexpr int BM = 128, BK = 32;
    constexpr int NT  = BN / 32;                    // B-tiles (16 cols) per wave
    constexpr int ACH = (BM * BK) / (256 * 8);      // 16B chunks per thread (A) = 2
    constexpr int BCH = (BN * BK) / (256 * 8);      // 2 (BN=128) or 1 (BN=64)

    __shared__ __align__(16) __bf16 As[2][BM * BK]; // [m][k]
    __shared__ __align__(16) __bf16 Bs[2][BN * BK]; // [n][k] (B pre-transposed)

    const int tid    = threadIdx.x;
    const int lane   = tid & 31;
    const int wave   = tid >> 5;
    const int waveM  = wave & 3;            // 4 waves along M (32 rows each)
    const int waveN  = wave >> 2;           // 2 waves along N (BN/2 cols each)
    const int blockM = blockIdx.y * BM;
    const int blockN = blockIdx.x * BN;
    const int lr     = lane & 15;
    const int hlf    = lane >> 4;

    // stage tile at k-offset kk into LDS buffer `buf`
    auto stage = [&](int buf, int kk) {
        if (FUSED_PHI) {
            // compute A tile: 16 elems per thread (4 groups of 4 along k)
#pragma unroll
            for (int t = 0; t < 4; ++t) {
                int elem = (t * 256 + tid) * 4;          // = m*32+k, k 4-aligned
                int m = elem >> 5, k = elem & 31;
                int gk = kk + k;                          // 4 elems share gk>>6
                float xv = (float)A[(size_t)(blockM + m) * D_ + (gk >> 6)];
                float4 w = *(const float4*)&Wphi[gk];
                float4 b = *(const float4*)&bphi[gk];
                v4bf p;
                p[0] = (__bf16)gelu_exact(xv * w.x + b.x);
                p[1] = (__bf16)gelu_exact(xv * w.y + b.y);
                p[2] = (__bf16)gelu_exact(xv * w.z + b.z);
                p[3] = (__bf16)gelu_exact(xv * w.w + b.w);
                *(v4bf*)&As[buf][elem] = p;               // 8B LDS store
            }
        } else {
#pragma unroll
            for (int t = 0; t < ACH; ++t) {
                int elem = (t * 256 + tid) * 8;           // 16B chunk, in-row
                int m = elem >> 5, k = elem & 31;
                async_copy_b128((unsigned)(uintptr_t)&As[buf][elem],
                                A + (size_t)(blockM + m) * K + kk + k);
            }
        }
#pragma unroll
        for (int t = 0; t < BCH; ++t) {
            int elem = (t * 256 + tid) * 8;
            int n = elem >> 5, k = elem & 31;
            async_copy_b128((unsigned)(uintptr_t)&Bs[buf][elem],
                            Bt + (size_t)(blockN + n) * K + kk + k);
        }
    };

    v8f acc[2][NT];
#pragma unroll
    for (int i = 0; i < 2; ++i)
#pragma unroll
        for (int j = 0; j < NT; ++j)
#pragma unroll
            for (int v = 0; v < 8; ++v) acc[i][j][v] = 0.0f;

    // prologue: stage first tile
    stage(0, 0);
    wait_async0();
    __syncthreads();

    int cur = 0;
    for (int k0 = 0; k0 < K; k0 += BK) {
        // overlap: issue async copies for the next tile into the other buffer
        if (k0 + BK < K) stage(cur ^ 1, k0 + BK);

        // ---- fragment loads (16B LDS vector reads matching WMMA layout) ----
        v16bf afrag[2], bfrag[NT];
#pragma unroll
        for (int i = 0; i < 2; ++i) {
            int row = waveM * 32 + i * 16 + lr;
            v8bf lo = *(const v8bf*)&As[cur][row * BK + hlf * 8];
            v8bf hi = *(const v8bf*)&As[cur][row * BK + 16 + hlf * 8];
#pragma unroll
            for (int t = 0; t < 8; ++t) { afrag[i][t] = lo[t]; afrag[i][t + 8] = hi[t]; }
        }
#pragma unroll
        for (int j = 0; j < NT; ++j) {
            int col = waveN * (BN / 2) + j * 16 + lr;
            v8bf lo = *(const v8bf*)&Bs[cur][col * BK + hlf * 8];
            v8bf hi = *(const v8bf*)&Bs[cur][col * BK + 16 + hlf * 8];
#pragma unroll
            for (int t = 0; t < 8; ++t) { bfrag[j][t] = lo[t]; bfrag[j][t + 8] = hi[t]; }
        }

#pragma unroll
        for (int i = 0; i < 2; ++i)
#pragma unroll
            for (int j = 0; j < NT; ++j)
                acc[i][j] = __builtin_amdgcn_wmma_f32_16x16x32_bf16(
                    false, afrag[i], false, bfrag[j], (short)0, acc[i][j],
                    false, false);

        wait_async0();       // next tile's copies have landed
        __syncthreads();     // all waves done reading `cur`, copies visible
        cur ^= 1;
    }

    // ---- epilogue: bias (+ optional exact GELU), f32 or bf16 store ----
#pragma unroll
    for (int i = 0; i < 2; ++i) {
#pragma unroll
        for (int j = 0; j < NT; ++j) {
            int col = blockN + waveN * (BN / 2) + j * 16 + lr;
            float bv = bias[col];
#pragma unroll
            for (int v = 0; v < 8; ++v) {
                int row = blockM + waveM * 32 + i * 16 + hlf * 8 + v;
                float o = acc[i][j][v] + bv;
                if (ACT == 1) o = gelu_exact(o);
                if (OUT_BF16) ((__bf16*)Cv)[(size_t)row * N + col] = (__bf16)o;
                else          ((float*)Cv)[(size_t)row * N + col] = o;
            }
        }
    }
}

// ---------------------------------------------------------------------------
// f32 -> bf16 convert (vectorized), n4 = total/4
// ---------------------------------------------------------------------------
__global__ void convert_bf16_kernel(const float* __restrict__ in, __bf16* __restrict__ out,
                                    size_t n4)
{
    size_t i = (size_t)blockIdx.x * 256 + threadIdx.x;
    if (i >= n4) return;
    float4 v = ((const float4*)in)[i];
    v4bf p;
    p[0] = (__bf16)v.x; p[1] = (__bf16)v.y; p[2] = (__bf16)v.z; p[3] = (__bf16)v.w;
    ((v4bf*)out)[i] = p;
}

// ---------------------------------------------------------------------------
// f32 [K,N] -> bf16 transposed [N,K]
// ---------------------------------------------------------------------------
__global__ void transpose_bf16_kernel(const float* __restrict__ in, __bf16* __restrict__ out,
                                      int K, int N)
{
    size_t i = (size_t)blockIdx.x * 256 + threadIdx.x;
    if (i >= (size_t)K * N) return;
    int k = (int)(i / N), n = (int)(i % N);
    out[(size_t)n * K + k] = (__bf16)in[i];
}

// ---------------------------------------------------------------------------
// am[c] = mean over rows of in[r, c]   (one block per column)
// ---------------------------------------------------------------------------
__global__ void colmean_kernel(const float* __restrict__ in, float* __restrict__ out,
                               int rows, int cols)
{
    __shared__ float red[256];
    int c = blockIdx.x;
    float s = 0.0f;
    for (int r = threadIdx.x; r < rows; r += 256) s += in[(size_t)r * cols + c];
    red[threadIdx.x] = s;
    __syncthreads();
    for (int off = 128; off > 0; off >>= 1) {
        if (threadIdx.x < off) red[threadIdx.x] += red[threadIdx.x + off];
        __syncthreads();
    }
    if (threadIdx.x == 0) out[c] = red[0] / (float)rows;
}

// ---------------------------------------------------------------------------
// Wdyn^T (bf16) builder: i = k*256 + n over [256,256] weight;
// outT[n*256+k] = Wb[i] + (am @ Wa)[i] + ba[i]
// ---------------------------------------------------------------------------
__global__ void wdyn_kernel(const float* __restrict__ Wb, const float* __restrict__ Wa,
                            const float* __restrict__ ba, const float* __restrict__ am,
                            __bf16* __restrict__ outT, int total)
{
    int i = blockIdx.x * 256 + threadIdx.x;
    if (i >= total) return;
    float acc = ba[i];
#pragma unroll 4
    for (int a = 0; a < ADAPT_; ++a) acc += am[a] * Wa[(size_t)a * total + i];
    int k = i >> 8, n = i & 255;
    outT[(size_t)n * D_ + k] = (__bf16)(Wb[i] + acc);
}

// ---------------------------------------------------------------------------
// gate[n,:] = softmax(adapt[n,:] @ Wg + bg)   (one token per thread)
// ---------------------------------------------------------------------------
__global__ void gate_kernel(const float* __restrict__ adapt, const float* __restrict__ Wg,
                            const float* __restrict__ bg, float* __restrict__ gate,
                            int n_tok)
{
    int n = blockIdx.x * 256 + threadIdx.x;
    if (n >= n_tok) return;
    float lg[NEXP_ + 1];
#pragma unroll
    for (int j = 0; j <= NEXP_; ++j) lg[j] = bg[j];
    for (int a = 0; a < ADAPT_; ++a) {
        float av = adapt[n * ADAPT_ + a];
#pragma unroll
        for (int j = 0; j <= NEXP_; ++j) lg[j] += av * Wg[a * (NEXP_ + 1) + j];
    }
    float mx = lg[0];
#pragma unroll
    for (int j = 1; j <= NEXP_; ++j) mx = fmaxf(mx, lg[j]);
    float ssum = 0.0f;
#pragma unroll
    for (int j = 0; j <= NEXP_; ++j) { lg[j] = __expf(lg[j] - mx); ssum += lg[j]; }
    float inv = 1.0f / ssum;
#pragma unroll
    for (int j = 0; j <= NEXP_; ++j) gate[n * (NEXP_ + 1) + j] = lg[j] * inv;
}

// ---------------------------------------------------------------------------
// concat[n,:] = [tok | chan | moe | lf]; writes f32 (for combine) + bf16 (GEMM A)
// ---------------------------------------------------------------------------
__global__ void concat_kernel(const float* __restrict__ tok, const float* __restrict__ chan,
                              const float* __restrict__ eo, const float* __restrict__ gate,
                              const float* __restrict__ ka, const float* __restrict__ lf,
                              float* __restrict__ cat, __bf16* __restrict__ catb, int n_tok)
{
    size_t i = (size_t)blockIdx.x * 256 + threadIdx.x;
    if (i >= (size_t)n_tok * DSUM_) return;
    int n = (int)(i / DSUM_);
    int c = (int)(i % DSUM_);
    float v;
    if (c < D_) {
        v = tok[n * D_ + c];
    } else if (c < 2 * D_) {
        v = chan[n * D_ + (c - D_)];
    } else if (c < 3 * D_) {
        int d = c - 2 * D_;
        float s = gate[n * (NEXP_ + 1) + NEXP_] * ka[n * D_ + d];
#pragma unroll
        for (int e = 0; e < NEXP_; ++e)
            s += gate[n * (NEXP_ + 1) + e] * eo[((size_t)e * n_tok + n) * D_ + d];
        v = s;
    } else {
        v = lf[(size_t)n * DLF_ + (c - 3 * D_)];
    }
    cat[i]  = v;
    catb[i] = (__bf16)v;
}

// ---------------------------------------------------------------------------
// cw[n,:] = softmax(h[n,:] @ W2 + b2)   (one token per thread)
// ---------------------------------------------------------------------------
__global__ void cw_kernel(const float* __restrict__ h, const float* __restrict__ W2,
                          const float* __restrict__ b2, float* __restrict__ cw, int n_tok)
{
    int n = blockIdx.x * 256 + threadIdx.x;
    if (n >= n_tok) return;
    float lg[4] = {b2[0], b2[1], b2[2], b2[3]};
    for (int k = 0; k < HID_; ++k) {
        float hv = h[n * HID_ + k];
#pragma unroll
        for (int j = 0; j < 4; ++j) lg[j] += hv * W2[k * 4 + j];
    }
    float mx = fmaxf(fmaxf(lg[0], lg[1]), fmaxf(lg[2], lg[3]));
    float ssum = 0.0f;
#pragma unroll
    for (int j = 0; j < 4; ++j) { lg[j] = __expf(lg[j] - mx); ssum += lg[j]; }
    float inv = 1.0f / ssum;
#pragma unroll
    for (int j = 0; j < 4; ++j) cw[n * 4 + j] = lg[j] * inv;
}

// ---------------------------------------------------------------------------
// combined (bf16, feeds final GEMM): cw[n,seg(c)] * concat[i] + residual[i]
// ---------------------------------------------------------------------------
__global__ void combined_kernel(const float* __restrict__ cat, const float* __restrict__ res,
                                const float* __restrict__ cw, __bf16* __restrict__ combb,
                                int n_tok)
{
    size_t i = (size_t)blockIdx.x * 256 + threadIdx.x;
    if (i >= (size_t)n_tok * DSUM_) return;
    int n = (int)(i / DSUM_);
    int c = (int)(i % DSUM_);
    int s = (c < D_) ? 0 : (c < 2 * D_) ? 1 : (c < 3 * D_) ? 2 : 3;
    combb[i] = (__bf16)(cw[n * 4 + s] * cat[i] + res[i]);
}

// ---------------------------------------------------------------------------
// Host-side launch sequence
// ---------------------------------------------------------------------------
extern "C" void kernel_launch(void* const* d_in, const int* in_sizes, int n_in,
                              void* d_out, int out_size, void* d_ws, size_t ws_size,
                              hipStream_t stream)
{
    (void)in_sizes; (void)n_in; (void)out_size; (void)ws_size;

    const float* x      = (const float*)d_in[0];
    const float* lf_out = (const float*)d_in[1];
    const float* W_feat = (const float*)d_in[2];
    const float* b_feat = (const float*)d_in[3];
    const float* Wt     = (const float*)d_in[4];
    const float* bt     = (const float*)d_in[5];
    const float* Wt_a   = (const float*)d_in[6];
    const float* bt_a   = (const float*)d_in[7];
    const float* Wc     = (const float*)d_in[8];
    const float* bc     = (const float*)d_in[9];
    const float* Wc_a   = (const float*)d_in[10];
    const float* bc_a   = (const float*)d_in[11];
    const float* We     = (const float*)d_in[12];
    const float* be     = (const float*)d_in[13];
    const float* We_a   = (const float*)d_in[14];
    const float* be_a   = (const float*)d_in[15];
    const float* Wg     = (const float*)d_in[16];
    const float* bg     = (const float*)d_in[17];
    const float* Wphi   = (const float*)d_in[18];
    const float* bphi   = (const float*)d_in[19];
    const float* Wpsi1  = (const float*)d_in[20];
    const float* bpsi1  = (const float*)d_in[21];
    const float* Wpsi2  = (const float*)d_in[22];
    const float* bpsi2  = (const float*)d_in[23];
    const float* Wres   = (const float*)d_in[24];
    const float* bres   = (const float*)d_in[25];
    const float* W1     = (const float*)d_in[26];
    const float* b1     = (const float*)d_in[27];
    const float* W2     = (const float*)d_in[28];
    const float* b2     = (const float*)d_in[29];
    const float* Wo     = (const float*)d_in[30];
    const float* bo     = (const float*)d_in[31];
    float* out = (float*)d_out;

    // ---- workspace layout: f32 region, then bf16 region ----
    float* ws = (float*)d_ws;
    float* adapt = ws;                                        // 4096*128
    float* am    = adapt + (size_t)NTOK * ADAPT_;             // 128
    float* tok   = am    + ADAPT_;                            // 4096*256
    float* chan  = tok   + (size_t)NTOK * D_;                 // 4096*256
    float* eo    = chan  + (size_t)NTOK * D_;                 // 4*4096*256
    float* gate  = eo    + (size_t)NEXP_ * NTOK * D_;         // 4096*5
    float* ka    = gate  + (size_t)NTOK * (NEXP_ + 1);        // 4096*256
    float* cat   = ka    + (size_t)NTOK * D_;                 // 4096*1536
    float* resid = cat   + (size_t)NTOK * DSUM_;              // 4096*1536
    float* hbuf  = resid + (size_t)NTOK * DSUM_;              // 4096*64
    float* cw    = hbuf  + (size_t)NTOK * HID_;               // 4096*4
    float* f32end = cw   + (size_t)NTOK * 4;

    __bf16* xb    = (__bf16*)f32end;                          // 4096*256
    __bf16* WfT   = xb    + (size_t)NTOK * D_;                // 128*256 (T)
    __bf16* wdT   = WfT   + (size_t)ADAPT_ * D_;              // 256*256 (T)
    __bf16* wcT   = wdT   + (size_t)D_ * D_;                  // 256*256 (T)
    __bf16* weT   = wcT   + (size_t)D_ * D_;                  // 4*256*256 (T)
    __bf16* Wp1T  = weT   + (size_t)NEXP_ * D_ * D_;          // 64*16384 (T)
    __bf16* Wp2T  = Wp1T  + (size_t)D_ * HID_ * HID_;         // 256*64 (T)
    __bf16* WresT = Wp2T  + (size_t)HID_ * D_;                // 1536*1536 (T)
    __bf16* W1T   = WresT + (size_t)DSUM_ * DSUM_;            // 64*1536 (T)
    __bf16* WoT   = W1T   + (size_t)DSUM_ * HID_;             // 256*1536 (T)
    __bf16* catb  = WoT   + (size_t)DSUM_ * D_;               // 4096*1536
    __bf16* combb = catb  + (size_t)NTOK * DSUM_;             // 4096*1536
    __bf16* hpsib = combb + (size_t)NTOK * DSUM_;             // 4096*64

    const dim3 blk(256);
    const int rowBlocks = NTOK / 128;   // 32

    // ---- one-time operand conversion / transposition (bandwidth-trivial) ----
    convert_bf16_kernel<<<dim3((NTOK * D_ / 4) / 256), blk, 0, stream>>>(
        x, xb, (size_t)NTOK * D_ / 4);
    transpose_bf16_kernel<<<dim3((D_ * ADAPT_) / 256), blk, 0, stream>>>(W_feat, WfT, D_, ADAPT_);
    transpose_bf16_kernel<<<dim3((D_ * HID_ * HID_) / 256), blk, 0, stream>>>(Wpsi1, Wp1T, D_ * HID_, HID_);
    transpose_bf16_kernel<<<dim3((HID_ * D_) / 256), blk, 0, stream>>>(Wpsi2, Wp2T, HID_, D_);
    transpose_bf16_kernel<<<dim3((DSUM_ * DSUM_) / 256), blk, 0, stream>>>(Wres, WresT, DSUM_, DSUM_);
    transpose_bf16_kernel<<<dim3((DSUM_ * HID_) / 256), blk, 0, stream>>>(W1, W1T, DSUM_, HID_);
    transpose_bf16_kernel<<<dim3((DSUM_ * D_) / 256), blk, 0, stream>>>(Wo, WoT, DSUM_, D_);

    // 1) adapt = x @ W_feat + b_feat     [4096,128]
    gemm_wmma_bf16<0, false, 128, false><<<dim3(1, rowBlocks), blk, 0, stream>>>(
        xb, WfT, b_feat, adapt, NTOK, ADAPT_, D_, nullptr, nullptr);

    // 2) am = mean over tokens
    colmean_kernel<<<dim3(ADAPT_), blk, 0, stream>>>(adapt, am, NTOK, ADAPT_);

    // 3) dynamic weights (written bf16, transposed) for mixers + 4 experts
    wdyn_kernel<<<dim3((D_ * D_) / 256), blk, 0, stream>>>(Wt, Wt_a, bt_a, am, wdT, D_ * D_);
    wdyn_kernel<<<dim3((D_ * D_) / 256), blk, 0, stream>>>(Wc, Wc_a, bc_a, am, wcT, D_ * D_);
    for (int e = 0; e < NEXP_; ++e) {
        wdyn_kernel<<<dim3((D_ * D_) / 256), blk, 0, stream>>>(
            We + (size_t)e * D_ * D_,
            We_a + (size_t)e * ADAPT_ * D_ * D_,
            be_a + (size_t)e * D_ * D_,
            am, weT + (size_t)e * D_ * D_, D_ * D_);
    }

    // 4) token / channel mixers   [4096,256]
    gemm_wmma_bf16<0, false, 128, false><<<dim3(D_ / 128, rowBlocks), blk, 0, stream>>>(
        xb, wdT, bt, tok, NTOK, D_, D_, nullptr, nullptr);
    gemm_wmma_bf16<0, false, 128, false><<<dim3(D_ / 128, rowBlocks), blk, 0, stream>>>(
        xb, wcT, bc, chan, NTOK, D_, D_, nullptr, nullptr);

    // 5) expert outputs eo[e]   [4096,256] each
    for (int e = 0; e < NEXP_; ++e) {
        gemm_wmma_bf16<0, false, 128, false><<<dim3(D_ / 128, rowBlocks), blk, 0, stream>>>(
            xb, weT + (size_t)e * D_ * D_, be + (size_t)e * D_,
            eo + (size_t)e * NTOK * D_, NTOK, D_, D_, nullptr, nullptr);
    }

    // 6) gating softmax
    gate_kernel<<<dim3(NTOK / 256), blk, 0, stream>>>(adapt, Wg, bg, gate, NTOK);

    // 7) KA expert: hpsi = gelu(phi_flat @ Wpsi1 + bpsi1), phi fused into A tile
    gemm_wmma_bf16<1, true, 64, true><<<dim3(1, rowBlocks), blk, 0, stream>>>(
        xb, Wp1T, bpsi1, hpsib, NTOK, HID_, D_ * HID_, Wphi, bphi);
    //    ka = hpsi @ Wpsi2 + bpsi2   [4096,256]
    gemm_wmma_bf16<0, false, 128, false><<<dim3(D_ / 128, rowBlocks), blk, 0, stream>>>(
        hpsib, Wp2T, bpsi2, ka, NTOK, D_, HID_, nullptr, nullptr);

    // 8) MoE combine + concat [tok|chan|moe|lf]   [4096,1536]  (f32 + bf16)
    concat_kernel<<<dim3((NTOK * DSUM_) / 256), blk, 0, stream>>>(
        tok, chan, eo, gate, ka, lf_out, cat, catb, NTOK);

    // 9) residual = concat @ Wres + bres   [4096,1536]  (dominant GEMM)
    gemm_wmma_bf16<0, false, 128, false><<<dim3(DSUM_ / 128, rowBlocks), blk, 0, stream>>>(
        catb, WresT, bres, resid, NTOK, DSUM_, DSUM_, nullptr, nullptr);

    // 10) h = gelu(concat @ W1 + b1)   [4096,64]
    gemm_wmma_bf16<1, false, 64, false><<<dim3(1, rowBlocks), blk, 0, stream>>>(
        catb, W1T, b1, hbuf, NTOK, HID_, DSUM_, nullptr, nullptr);

    // 11) component weights
    cw_kernel<<<dim3(NTOK / 256), blk, 0, stream>>>(hbuf, W2, b2, cw, NTOK);

    // 12) combined (bf16) = cw-scaled concat + residual
    combined_kernel<<<dim3((NTOK * DSUM_) / 256), blk, 0, stream>>>(
        cat, resid, cw, combb, NTOK);

    // 13) out = combined @ Wo + bo   [4096,256]
    gemm_wmma_bf16<0, false, 128, false><<<dim3(D_ / 128, rowBlocks), blk, 0, stream>>>(
        combb, WoT, bo, out, NTOK, D_, DSUM_, nullptr, nullptr);
}
